// DeepSpeedCompatibleMultiheadAttention_41171556499592
// MI455X (gfx1250) — matrix operand order repaired and checked by reading, not compiled, not run on hardware
//
#include <hip/hip_runtime.h>
#include <hip/hip_bf16.h>

typedef unsigned short u16;
typedef __attribute__((ext_vector_type(16))) __bf16 bf16x16;
typedef __attribute__((ext_vector_type(8)))  float          v8f;
typedef __attribute__((ext_vector_type(8)))  unsigned short u16x8;

constexpr int BB = 2, SS = 2048, EE = 1024, HH = 16, DH = 64;
constexpr int MM = BB * SS;            // 4096 rows of activations
constexpr float SCALE = 0.125f;        // 1/sqrt(64)

// ---------- helpers ----------

__device__ inline u16 f2bf(float x) {            // round-to-nearest-even fp32->bf16
  unsigned u = __float_as_uint(x);
  u += 0x7FFFu + ((u >> 16) & 1u);
  return (u16)(u >> 16);
}

__device__ inline v8f zero8() {
  v8f z = {0.f, 0.f, 0.f, 0.f, 0.f, 0.f, 0.f, 0.f};
  return z;
}

__device__ inline v8f wmma_bf16(bf16x16 a, bf16x16 b, v8f c) {
  // D = A(16x32 bf16) * B(32x16 bf16) + C(16x16 f32)
  return __builtin_amdgcn_wmma_f32_16x16x32_bf16(false, a, false, b, (short)0, c,
                                                 false, false);
}

// A fragment (16x32, MxK): lane&15 = M row, hi = lane>>4.
// element e -> K = 16*(e>>3) + 8*hi + (e&7): two contiguous 8-elt chunks.
__device__ inline bf16x16 load_a_frag(const u16* __restrict__ A, int lda,
                                      int row0, int k0, int lane) {
  const int row = row0 + (lane & 15);
  const int hi  = lane >> 4;
  const u16* p  = A + (size_t)row * lda + k0 + hi * 8;
  union { bf16x16 v; u16x8 h[2]; } u;
  u.h[0] = *(const u16x8*)(p);
  u.h[1] = *(const u16x8*)(p + 16);
  return u.v;
}

// B fragment (32x16, KxN): lane&15 = N col, element e -> K = 16*(e>>3) + 8*hi + (e&7)
// => two contiguous 8-elt chunks per lane (two 16B loads).
__device__ inline bf16x16 load_b_frag(const u16* __restrict__ Wm, int ldw,
                                      int col0, int k0, int lane) {
  const int col = col0 + (lane & 15);
  const int hi  = lane >> 4;
  const u16* p  = Wm + (size_t)col * ldw + k0 + hi * 8;
  union { bf16x16 v; u16x8 h[2]; } u;
  u.h[0] = *(const u16x8*)(p);
  u.h[1] = *(const u16x8*)(p + 16);
  return u.v;
}

// ---------- kernels ----------

__global__ void f32_to_bf16_kernel(const float* __restrict__ src,
                                   u16* __restrict__ dst, int n) {
  int i = blockIdx.x * blockDim.x + threadIdx.x;
  const int stride = gridDim.x * blockDim.x;
  for (; i < n; i += stride) dst[i] = f2bf(src[i]);
}

// Y[m,n] = sum_k A[m,k] * W[n,k] + bias[n]   (A: [4096,1024] bf16, W: [1024,1024] bf16)
// mode 0: store bf16 at [B,H,S,D]   (Q, K)
// mode 1: store bf16 at [B,H,D,S]   (V, transposed for PV fragment loads)
// mode 2: store fp32 at [m,n]       (final output projection)
__global__ __launch_bounds__(128)
void gemm_bf16_wmma_kernel(const u16* __restrict__ A,
                           const u16* __restrict__ Wt,
                           const float* __restrict__ bias,
                           void* __restrict__ out, int mode) {
  const int lane = threadIdx.x & 31;
  const int wv   = threadIdx.x >> 5;
  const int tile = blockIdx.x * 4 + wv;      // 1024 wave tiles of 64x64
  const int m0   = (tile / (EE / 64)) * 64;
  const int n0   = (tile % (EE / 64)) * 64;
  const int hi   = lane >> 4;

  v8f acc[4][4];
#pragma unroll
  for (int i = 0; i < 4; ++i)
#pragma unroll
    for (int j = 0; j < 4; ++j) acc[i][j] = zero8();

  for (int k0 = 0; k0 < EE; k0 += 32) {
    bf16x16 af[4], bfr[4];
#pragma unroll
    for (int i = 0; i < 4; ++i) af[i]  = load_a_frag(A,  EE, m0 + 16 * i, k0, lane);
#pragma unroll
    for (int j = 0; j < 4; ++j) bfr[j] = load_b_frag(Wt, EE, n0 + 16 * j, k0, lane);
#pragma unroll
    for (int i = 0; i < 4; ++i)
#pragma unroll
      for (int j = 0; j < 4; ++j) acc[i][j] = wmma_bf16(af[i], bfr[j], acc[i][j]);
  }

#pragma unroll
  for (int i = 0; i < 4; ++i) {
#pragma unroll
    for (int j = 0; j < 4; ++j) {
      const int n  = n0 + 16 * j + (lane & 15);
      const float bv = bias[n];
#pragma unroll
      for (int e = 0; e < 8; ++e) {
        const int m = m0 + 16 * i + 8 * hi + e;   // C/D layout: row = e + 8*hi
        const float val = acc[i][j][e] + bv;
        if (mode == 2) {
          ((float*)out)[(size_t)m * EE + n] = val;
        } else {
          const int b = m >> 11, s = m & (SS - 1);
          const int h = n >> 6,  d = n & (DH - 1);
          u16* o16 = (u16*)out;
          if (mode == 0)
            o16[((size_t)(b * HH + h) * SS + s) * DH + d] = f2bf(val);
          else
            o16[((size_t)(b * HH + h) * DH + d) * SS + s] = f2bf(val);
        }
      }
    }
  }
}

// Flash attention: one wave handles 16 query rows for one (b,h).
// Q: [B,H,S,D] bf16, K: [B,H,S,D] bf16, V: [B,H,D,S] bf16 (transposed)
// attn out: [B,S,E] bf16
__global__ __launch_bounds__(128)
void flash_attn_kernel(const u16* __restrict__ Qb, const u16* __restrict__ Kb,
                       const u16* __restrict__ Vt, u16* __restrict__ attn) {
  __shared__ __attribute__((aligned(16))) u16 Pbuf[4][16 * 32];

  const int lane = threadIdx.x & 31;
  const int wv   = threadIdx.x >> 5;
  const int wid  = blockIdx.x * 4 + wv;          // 4096 waves total
  const int qblk = wid & (SS / 16 - 1);
  const int h    = (wid >> 7) & (HH - 1);
  const int b    = wid >> 11;
  const int hi   = lane >> 4;
  const int col  = lane & 15;

  const u16* Qbase = Qb + (size_t)(b * HH + h) * SS * DH;
  const u16* Kbase = Kb + (size_t)(b * HH + h) * SS * DH;
  const u16* Vbase = Vt + (size_t)(b * HH + h) * DH * SS;
  const int q0 = qblk * 16;

  bf16x16 qa[2];
  qa[0] = load_a_frag(Qbase, DH, q0, 0,  lane);
  qa[1] = load_a_frag(Qbase, DH, q0, 32, lane);

  float mrow[8], lrow[8];
#pragma unroll
  for (int e = 0; e < 8; ++e) { mrow[e] = -1e30f; lrow[e] = 0.f; }
  v8f o[4];
#pragma unroll
  for (int j = 0; j < 4; ++j) o[j] = zero8();

  for (int kt = 0; kt < SS; kt += 32) {
    // scores: 16 queries x 32 keys, contraction over D=64 (two chained WMMAs)
    v8f s0 = zero8(), s1 = zero8();
#pragma unroll
    for (int c = 0; c < 2; ++c) {
      bf16x16 kb0 = load_b_frag(Kbase, DH, kt,      32 * c, lane);
      bf16x16 kb1 = load_b_frag(Kbase, DH, kt + 16, 32 * c, lane);
      s0 = wmma_bf16(qa[c], kb0, s0);
      s1 = wmma_bf16(qa[c], kb1, s1);
    }

    // online softmax over key dimension; rows live in the two 16-lane halves
    float alpha[8];
#pragma unroll
    for (int e = 0; e < 8; ++e) {
      const float a0 = s0[e] * SCALE, a1 = s1[e] * SCALE;
      float cur = fmaxf(a0, a1);
#pragma unroll
      for (int off = 1; off < 16; off <<= 1)
        cur = fmaxf(cur, __shfl_xor(cur, off, 32));
      const float mn = fmaxf(mrow[e], cur);
      alpha[e] = __expf(mrow[e] - mn);
      mrow[e]  = mn;
      const float p0 = __expf(a0 - mn);
      const float p1 = __expf(a1 - mn);
      float rs = p0 + p1;
#pragma unroll
      for (int off = 1; off < 16; off <<= 1) rs += __shfl_xor(rs, off, 32);
      lrow[e] = lrow[e] * alpha[e] + rs;
      const int r = e + 8 * hi;                  // C-frag row -> logical q row
      Pbuf[wv][r * 32 + col]      = f2bf(p0);
      Pbuf[wv][r * 32 + 16 + col] = f2bf(p1);
    }
    asm volatile("s_wait_dscnt 0x0" ::: "memory");   // LDS transpose round-trip

    // reload P in A-fragment layout (16 q x 32 keys)
    union { bf16x16 v; u16x8 hh[2]; } pu;
    {
      const u16* p = &Pbuf[wv][col * 32 + hi * 8];
      pu.hh[0] = *(const u16x8*)(p);
      pu.hh[1] = *(const u16x8*)(p + 16);
    }

    // O = O*alpha + P@V  (V transposed: contiguous over key index)
#pragma unroll
    for (int j = 0; j < 4; ++j) {
#pragma unroll
      for (int e = 0; e < 8; ++e) o[j][e] *= alpha[e];
      bf16x16 vb = load_b_frag(Vbase, SS, 16 * j, kt, lane);
      o[j] = wmma_bf16(pu.v, vb, o[j]);
    }
  }

  // normalize and store bf16 at [B,S,E] (fused un-transpose of heads)
#pragma unroll
  for (int j = 0; j < 4; ++j) {
#pragma unroll
    for (int e = 0; e < 8; ++e) {
      const float val = o[j][e] / lrow[e];
      const int q = q0 + e + 8 * hi;
      attn[((size_t)(b * SS + q)) * EE + h * DH + 16 * j + col] = f2bf(val);
    }
  }
}

// ---------- host ----------

extern "C" void kernel_launch(void* const* d_in, const int* in_sizes, int n_in,
                              void* d_out, int out_size, void* d_ws, size_t ws_size,
                              hipStream_t stream) {
  const float* q_in = (const float*)d_in[0];
  const float* k_in = (const float*)d_in[1];
  const float* v_in = (const float*)d_in[2];
  const float* Wq = (const float*)d_in[3];  const float* bq = (const float*)d_in[4];
  const float* Wk = (const float*)d_in[5];  const float* bk = (const float*)d_in[6];
  const float* Wv = (const float*)d_in[7];  const float* bv = (const float*)d_in[8];
  const float* Wo = (const float*)d_in[9];  const float* bo = (const float*)d_in[10];

  const size_t actN = (size_t)MM * EE;   // 4M elems
  const size_t wN   = (size_t)EE * EE;   // 1M elems

  char* ws = (char*)d_ws;
  u16* qa_bf = (u16*)ws; ws += actN * 2;
  u16* ka_bf = (u16*)ws; ws += actN * 2;
  u16* va_bf = (u16*)ws; ws += actN * 2;
  u16* Wq_bf = (u16*)ws; ws += wN * 2;
  u16* Wk_bf = (u16*)ws; ws += wN * 2;
  u16* Wv_bf = (u16*)ws; ws += wN * 2;
  u16* Wo_bf = (u16*)ws; ws += wN * 2;
  u16* Qh    = (u16*)ws; ws += actN * 2;   // [B,H,S,D]
  u16* Kh    = (u16*)ws; ws += actN * 2;   // [B,H,S,D]
  u16* Vth   = (u16*)ws; ws += actN * 2;   // [B,H,D,S]
  u16* At    = (u16*)ws; ws += actN * 2;   // attention out [B,S,E] bf16

  f32_to_bf16_kernel<<<2048, 256, 0, stream>>>(q_in, qa_bf, (int)actN);
  f32_to_bf16_kernel<<<2048, 256, 0, stream>>>(k_in, ka_bf, (int)actN);
  f32_to_bf16_kernel<<<2048, 256, 0, stream>>>(v_in, va_bf, (int)actN);
  f32_to_bf16_kernel<<<1024, 256, 0, stream>>>(Wq, Wq_bf, (int)wN);
  f32_to_bf16_kernel<<<1024, 256, 0, stream>>>(Wk, Wk_bf, (int)wN);
  f32_to_bf16_kernel<<<1024, 256, 0, stream>>>(Wv, Wv_bf, (int)wN);
  f32_to_bf16_kernel<<<1024, 256, 0, stream>>>(Wo, Wo_bf, (int)wN);

  gemm_bf16_wmma_kernel<<<256, 128, 0, stream>>>(qa_bf, Wq_bf, bq, Qh,  0);
  gemm_bf16_wmma_kernel<<<256, 128, 0, stream>>>(ka_bf, Wk_bf, bk, Kh,  0);
  gemm_bf16_wmma_kernel<<<256, 128, 0, stream>>>(va_bf, Wv_bf, bv, Vth, 1);

  flash_attn_kernel<<<1024, 128, 0, stream>>>(Qh, Kh, Vth, At);

  gemm_bf16_wmma_kernel<<<256, 128, 0, stream>>>(At, Wo_bf, bo, d_out, 2);
}
